// GABlock_39195871543951
// MI455X (gfx1250) — matrix operand order repaired and checked by reading, not compiled, not run on hardware
//
#include <hip/hip_runtime.h>
#include <hip/hip_bf16.h>
#include <cstdint>

typedef _Float16 f16;
typedef __attribute__((ext_vector_type(16))) _Float16 v16h;
typedef __attribute__((ext_vector_type(8)))  float    v8f;

#define LL   384
#define FF   128
#define CC   64
#define HH   12
#define QKD  32
#define VD   32
#define PQN  8
#define PVN  8
#define DOUT 1824   // H*C(768) + H*V(384) + H*PV*3(288) + H*PV(96) + H*PV*3(288)
#define HQK  (HH*QKD)   // 384
#define HP3  (HH*PQN*3) // 288

#define GF_RELU   1
#define GF_OUTF16 2
#define GF_OUTF16T 4

// ---------------- WMMA tile loaders (wave32, v_wmma_f32_16x16x32_f16) ----------
// A: row-major M x K (k contiguous). ISA 16-bit A 16x32 layout:
// lane<16: m=lane, halves 0..7 = K 0..7, halves 8..15 = K16..23
// lane>=16: m=lane-16, K 8..15 and 24..31.
__device__ __forceinline__ v16h load_a16(const f16* A, int lda, int m0, int k0) {
  int lane = threadIdx.x & 31;
  const f16* p = A + (size_t)(m0 + (lane & 15)) * lda + k0 + ((lane < 16) ? 0 : 8);
  v16h r;
#pragma unroll
  for (int t = 0; t < 8; ++t) r[t] = p[t];
#pragma unroll
  for (int t = 0; t < 8; ++t) r[8 + t] = p[16 + t];
  return r;
}
__device__ __forceinline__ v16h load_a16_k(const f16* A, int lda, int m0, int klim) {
  int lane = threadIdx.x & 31;
  int kb = (lane < 16) ? 0 : 8;
  const f16* p = A + (size_t)(m0 + (lane & 15)) * lda;
  v16h r;
#pragma unroll
  for (int t = 0; t < 8; ++t) { int k = kb + t;      r[t]     = (k < klim) ? p[k] : (f16)0.f; }
#pragma unroll
  for (int t = 0; t < 8; ++t) { int k = kb + 16 + t; r[8 + t] = (k < klim) ? p[k] : (f16)0.f; }
  return r;
}
// B source is N x K row-major (we compute A * B^T). ISA B 32x16 layout:
// lanes0-15 hold K=0..7,16..23; lanes16-31 hold K=8..15,24..31 (same as A); N = lane&15.
__device__ __forceinline__ v16h load_b16(const f16* B, int ldb, int n0, int k0) {
  int lane = threadIdx.x & 31;
  const f16* p = B + (size_t)(n0 + (lane & 15)) * ldb + k0 + ((lane < 16) ? 0 : 8);
  v16h r;
#pragma unroll
  for (int t = 0; t < 16; ++t) r[t] = p[t + (t & 8)];
  return r;
}
__device__ __forceinline__ v16h load_b16_k(const f16* B, int ldb, int n0, int klim) {
  int lane = threadIdx.x & 31;
  int kb = (lane < 16) ? 0 : 8;
  const f16* p = B + (size_t)(n0 + (lane & 15)) * ldb;
  v16h r;
#pragma unroll
  for (int t = 0; t < 16; ++t) { int k = kb + t + (t & 8); r[t] = (k < klim) ? p[k] : (f16)0.f; }
  return r;
}

// -------- generic batched GEMM: C[bz][m][n] = scale * sum_k A[m][k]*B[n][k] + bias[n]
// one wave (32 threads) per 16x16 tile; K multiple of 32.
__global__ void k_gemm(const f16* __restrict__ A, int lda, long long sA,
                       const f16* __restrict__ B, int ldb, long long sB,
                       float* __restrict__ C, f16* __restrict__ C16, int ldc, long long sC,
                       int K, int Mlim, int Nlim,
                       const float* __restrict__ bias, float scale, int flags) {
  int m0 = blockIdx.x * 16, n0 = blockIdx.y * 16, bz = blockIdx.z;
  A += (long long)bz * sA;
  B += (long long)bz * sB;
  v8f acc = {};
  for (int k = 0; k < K; k += 32) {
    v16h a = load_a16(A, lda, m0, k);
    v16h b = load_b16(B, ldb, n0, k);
    acc = __builtin_amdgcn_wmma_f32_16x16x32_f16(false, a, false, b, (short)0, acc, false, false);
  }
  int lane = threadIdx.x & 31;
  int n = n0 + (lane & 15);
  if (n >= Nlim) return;
  float bv = bias ? bias[n] : 0.f;
  long long co = (long long)bz * sC;
#pragma unroll
  for (int r = 0; r < 8; ++r) {
    int m = m0 + r + ((lane < 16) ? 0 : 8);
    if (m >= Mlim) continue;
    float v = acc[r] * scale + bv;
    if (flags & GF_RELU) v = fmaxf(v, 0.f);
    if (flags & GF_OUTF16T)      C16[co + (long long)n * ldc + m] = (f16)v;
    else if (flags & GF_OUTF16)  C16[co + (long long)m * ldc + n] = (f16)v;
    else                         C[co + (long long)m * ldc + n] = v;
  }
}

// -------- fused logits: node WMMA (K=32) + spatial cross WMMA (K=24 padded) + rank-1 + pair + mask
__global__ void k_combine(const f16* __restrict__ q16s, const f16* __restrict__ k16,
                          const f16* __restrict__ qpg, const f16* __restrict__ kpg,
                          const float* __restrict__ pair, const float* __restrict__ qn,
                          const float* __restrict__ kn, const float* __restrict__ gcoef,
                          const unsigned char* __restrict__ mask, float* __restrict__ logits) {
  int i0 = blockIdx.x * 16, j0 = blockIdx.y * 16, h = blockIdx.z;
  v8f acc = {};
  v16h a = load_a16(q16s + h * QKD, HQK, i0, 0);
  v16h b = load_b16(k16 + h * QKD, HQK, j0, 0);
  acc = __builtin_amdgcn_wmma_f32_16x16x32_f16(false, a, false, b, (short)0, acc, false, false);
  a = load_a16_k(qpg + h * 24, HP3, i0, 24);
  b = load_b16_k(kpg + h * 24, HP3, j0, 24);
  acc = __builtin_amdgcn_wmma_f32_16x16x32_f16(false, a, false, b, (short)0, acc, false, false);
  int lane = threadIdx.x & 31;
  int n = j0 + (lane & 15);
  float knv = kn[n * HH + h];
  bool mj = mask[n] != 0;
  float gc = gcoef[h];
#pragma unroll
  for (int r = 0; r < 8; ++r) {
    int m = i0 + r + ((lane < 16) ? 0 : 8);
    float v = acc[r] + pair[((size_t)m * LL + n) * HH + h] - (qn[m * HH + h] + knv) * gc;
    if (!(mask[m] && mj)) v -= 100000.0f;
    logits[((size_t)h * LL + m) * LL + n] = v;
  }
}

// -------- elementwise helpers ----------
__global__ void k_conv16(const float* __restrict__ s, long long n, f16* __restrict__ d, float scale) {
  long long i = (long long)blockIdx.x * blockDim.x + threadIdx.x;
  if (i < n) d[i] = (f16)(s[i] * scale);
}
// src[R][C] f32 -> dst[Cpad][R] f16 (rows >= C zeroed)
__global__ void k_tr16(const float* __restrict__ s, int R, int C, f16* __restrict__ d, int Cpad) {
  int i = blockIdx.x * blockDim.x + threadIdx.x;
  if (i >= Cpad * R) return;
  int r = i % R, c = i / R;
  d[(size_t)c * R + r] = (c < C) ? (f16)s[(size_t)r * C + c] : (f16)0.f;
}
// z -> z16 (flat) and zt16 (per-i transposed [i][c][j])
__global__ void k_zconv(const float* __restrict__ z, f16* __restrict__ z16, f16* __restrict__ zt16) {
  long long idx = (long long)blockIdx.x * blockDim.x + threadIdx.x;
  if (idx >= (long long)LL * LL * CC) return;
  int c = idx & (CC - 1);
  long long ij = idx >> 6;
  int j = (int)(ij % LL), i = (int)(ij / LL);
  f16 v = (f16)z[idx];
  z16[idx] = v;
  zt16[((size_t)i * CC + c) * LL + j] = v;
}

// -------- rigid transform of q/k/v points; norms; per-head scales ----------
__global__ void k_points(const float* __restrict__ R, const float* __restrict__ t,
                         const float* __restrict__ qp, const float* __restrict__ kp,
                         const float* __restrict__ vp, const float* __restrict__ spc,
                         f16* __restrict__ qpg, f16* __restrict__ kpg, f16* __restrict__ vpt,
                         float* __restrict__ qn, float* __restrict__ kn, float* __restrict__ gcoef) {
  int idx = blockIdx.x * blockDim.x + threadIdx.x;
  if (idx >= LL * HH) return;
  int h = idx % HH, i = idx / HH;
  const float* Ri = R + i * 9;
  const float* ti = t + i * 3;
  float g = logf(1.f + expf(spc[h]));            // softplus
  const float c2 = 1.f / 12.f, is3 = 0.57735026919f;
  float sA = 2.f * g * c2 * is3;                 // A-side scale for cross term
  if (i == 0) gcoef[h] = g * c2 * is3;           // rank-1 coefficient
  float qa = 0.f, ka = 0.f;
#pragma unroll 1
  for (int p = 0; p < PQN; ++p) {
    int o = i * HP3 + h * 24 + p * 3;
    float px = qp[o], py = qp[o + 1], pz = qp[o + 2];
    float gx = Ri[0] * px + Ri[1] * py + Ri[2] * pz + ti[0];
    float gy = Ri[3] * px + Ri[4] * py + Ri[5] * pz + ti[1];
    float gz = Ri[6] * px + Ri[7] * py + Ri[8] * pz + ti[2];
    qa += gx * gx + gy * gy + gz * gz;
    qpg[o] = (f16)(gx * sA); qpg[o + 1] = (f16)(gy * sA); qpg[o + 2] = (f16)(gz * sA);
    px = kp[o]; py = kp[o + 1]; pz = kp[o + 2];
    gx = Ri[0] * px + Ri[1] * py + Ri[2] * pz + ti[0];
    gy = Ri[3] * px + Ri[4] * py + Ri[5] * pz + ti[1];
    gz = Ri[6] * px + Ri[7] * py + Ri[8] * pz + ti[2];
    ka += gx * gx + gy * gy + gz * gz;
    kpg[o] = (f16)gx; kpg[o + 1] = (f16)gy; kpg[o + 2] = (f16)gz;
    px = vp[o]; py = vp[o + 1]; pz = vp[o + 2];
    gx = Ri[0] * px + Ri[1] * py + Ri[2] * pz + ti[0];
    gy = Ri[3] * px + Ri[4] * py + Ri[5] * pz + ti[1];
    gz = Ri[6] * px + Ri[7] * py + Ri[8] * pz + ti[2];
    vpt[((size_t)h * 32 + p * 3 + 0) * LL + i] = (f16)gx;   // transposed for B-side
    vpt[((size_t)h * 32 + p * 3 + 1) * LL + i] = (f16)gy;
    vpt[((size_t)h * 32 + p * 3 + 2) * LL + i] = (f16)gz;
  }
  for (int pc = 24; pc < 32; ++pc) vpt[((size_t)h * 32 + pc) * LL + i] = (f16)0.f;
  qn[i * HH + h] = qa;
  kn[i * HH + h] = ka;
}

// -------- softmax over j (384) per (h,i); f32 math, f16 alpha out --------
__global__ void k_softmax(const float* __restrict__ logits, const unsigned char* __restrict__ mask,
                          f16* __restrict__ alpha16) {
  __shared__ float sh[128];
  int i = blockIdx.x, h = blockIdx.y, t = threadIdx.x;
  const float* row = logits + ((size_t)h * LL + i) * LL;
  float v0 = row[t], v1 = row[t + 128], v2 = row[t + 256];
  sh[t] = fmaxf(fmaxf(v0, v1), v2);
  __syncthreads();
  for (int s = 64; s > 0; s >>= 1) { if (t < s) sh[t] = fmaxf(sh[t], sh[t + s]); __syncthreads(); }
  float mx = sh[0];
  __syncthreads();
  float e0 = __expf(v0 - mx), e1 = __expf(v1 - mx), e2 = __expf(v2 - mx);
  sh[t] = e0 + e1 + e2;
  __syncthreads();
  for (int s = 64; s > 0; s >>= 1) { if (t < s) sh[t] += sh[t + s]; __syncthreads(); }
  float inv = 1.f / sh[0];
  bool mi = mask[i] != 0;
  f16* o = alpha16 + ((size_t)h * LL + i) * LL;
  o[t]       = mi ? (f16)(e0 * inv) : (f16)0.f;
  o[t + 128] = mi ? (f16)(e1 * inv) : (f16)0.f;
  o[t + 256] = mi ? (f16)(e2 * inv) : (f16)0.f;
}

// -------- spatial epilogue: local frame, dist, direction into feat_all --------
__global__ void k_spatial(const float* __restrict__ R, const float* __restrict__ t,
                          const float* __restrict__ aggr, float* __restrict__ featall) {
  int idx = blockIdx.x * blockDim.x + threadIdx.x;
  if (idx >= LL * HH * PVN) return;
  int p = idx & 7, h = (idx >> 3) % HH, i = idx / (HH * PVN);
  const float* Ri = R + i * 9;
  const float* ti = t + i * 3;
  int o = i * HP3 + h * 24 + p * 3;
  float gx = aggr[o] - ti[0], gy = aggr[o + 1] - ti[1], gz = aggr[o + 2] - ti[2];
  float lx = Ri[0] * gx + Ri[3] * gy + Ri[6] * gz;   // R^T (g - t)
  float ly = Ri[1] * gx + Ri[4] * gy + Ri[7] * gz;
  float lz = Ri[2] * gx + Ri[5] * gy + Ri[8] * gz;
  float d = sqrtf(lx * lx + ly * ly + lz * lz);
  float id = 1.f / (d + 1e-4f);
  float* fr = featall + (size_t)i * DOUT;
  fr[1152 + h * 24 + p * 3 + 0] = lx;
  fr[1152 + h * 24 + p * 3 + 1] = ly;
  fr[1152 + h * 24 + p * 3 + 2] = lz;
  fr[1440 + h * 8 + p] = d;
  fr[1536 + h * 24 + p * 3 + 0] = lx * id;
  fr[1536 + h * 24 + p * 3 + 1] = ly * id;
  fr[1536 + h * 24 + p * 3 + 2] = lz * id;
}

// -------- residual + LayerNorm (optionally mask-zeroes b, optionally emits f16) ----
__global__ void k_ln(const float* __restrict__ a, const float* __restrict__ b,
                     const unsigned char* __restrict__ mask,
                     const float* __restrict__ g, const float* __restrict__ be,
                     float* __restrict__ out, f16* __restrict__ out16) {
  __shared__ float sh[128];
  int row = blockIdx.x, t = threadIdx.x;
  float bv = b[row * FF + t];
  if (mask && !mask[row]) bv = 0.f;
  float v = a[row * FF + t] + bv;
  sh[t] = v;
  __syncthreads();
  for (int s = 64; s > 0; s >>= 1) { if (t < s) sh[t] += sh[t + s]; __syncthreads(); }
  float mean = sh[0] * (1.f / FF);
  __syncthreads();
  float d = v - mean;
  sh[t] = d * d;
  __syncthreads();
  for (int s = 64; s > 0; s >>= 1) { if (t < s) sh[t] += sh[t + s]; __syncthreads(); }
  float var = sh[0] * (1.f / FF);
  float r = d * rsqrtf(var + 1e-5f) * g[t] + be[t];
  out[row * FF + t] = r;
  if (out16) out16[row * FF + t] = (f16)r;
}

extern "C" void kernel_launch(void* const* d_in, const int* in_sizes, int n_in,
                              void* d_out, int out_size, void* d_ws, size_t ws_size,
                              hipStream_t stream) {
  const float* R    = (const float*)d_in[0];
  const float* t    = (const float*)d_in[1];
  const float* x    = (const float*)d_in[2];
  const float* z    = (const float*)d_in[3];
  const unsigned char* mask = (const unsigned char*)d_in[4];
  const float* Wq   = (const float*)d_in[5];
  const float* Wk   = (const float*)d_in[6];
  const float* Wv   = (const float*)d_in[7];
  const float* Wpb  = (const float*)d_in[8];
  const float* spc  = (const float*)d_in[9];
  const float* Wqp  = (const float*)d_in[10];
  const float* Wkp  = (const float*)d_in[11];
  const float* Wvp  = (const float*)d_in[12];
  const float* Wout = (const float*)d_in[13];
  const float* bout = (const float*)d_in[14];
  const float* ln1g = (const float*)d_in[15];
  const float* ln1b = (const float*)d_in[16];
  const float* W1   = (const float*)d_in[17];
  const float* b1   = (const float*)d_in[18];
  const float* W2   = (const float*)d_in[19];
  const float* b2   = (const float*)d_in[20];
  const float* W3   = (const float*)d_in[21];
  const float* b3   = (const float*)d_in[22];
  const float* ln2g = (const float*)d_in[23];
  const float* ln2b = (const float*)d_in[24];

  char* ws = (char*)d_ws;
  size_t off = 0;
  auto alloc = [&](size_t bytes) { void* p = ws + off; off = (off + bytes + 255) & ~(size_t)255; return p; };

  f16*   x16    = (f16*)alloc((size_t)LL * FF * 2);
  f16*   WqT    = (f16*)alloc((size_t)HQK * FF * 2);
  f16*   WkT    = (f16*)alloc((size_t)HQK * FF * 2);
  f16*   WvT    = (f16*)alloc((size_t)HQK * FF * 2);
  f16*   WqpT   = (f16*)alloc((size_t)HP3 * FF * 2);
  f16*   WkpT   = (f16*)alloc((size_t)HP3 * FF * 2);
  f16*   WvpT   = (f16*)alloc((size_t)HP3 * FF * 2);
  f16*   WpbT   = (f16*)alloc((size_t)16 * CC * 2);
  f16*   WoutT  = (f16*)alloc((size_t)FF * DOUT * 2);
  f16*   W1T    = (f16*)alloc((size_t)FF * FF * 2);
  f16*   W2T    = (f16*)alloc((size_t)FF * FF * 2);
  f16*   W3T    = (f16*)alloc((size_t)FF * FF * 2);
  f16*   q16s   = (f16*)alloc((size_t)LL * HQK * 2);
  f16*   k16    = (f16*)alloc((size_t)LL * HQK * 2);
  f16*   vt16   = (f16*)alloc((size_t)HQK * LL * 2);        // [o=h*32+d][i]
  float* qp_l   = (float*)alloc((size_t)LL * HP3 * 4);
  float* kp_l   = (float*)alloc((size_t)LL * HP3 * 4);
  float* vp_l   = (float*)alloc((size_t)LL * HP3 * 4);
  f16*   qpg    = (f16*)alloc((size_t)LL * HP3 * 2);
  f16*   kpg    = (f16*)alloc((size_t)LL * HP3 * 2);
  f16*   vpt16  = (f16*)alloc((size_t)HH * 32 * LL * 2);    // [h][pc(pad32)][i]
  float* qn     = (float*)alloc((size_t)LL * HH * 4);
  float* kn     = (float*)alloc((size_t)LL * HH * 4);
  float* gcoef  = (float*)alloc(16 * 4);
  f16*   z16    = (f16*)alloc((size_t)LL * LL * CC * 2);
  f16*   zt16   = (f16*)alloc((size_t)LL * CC * LL * 2);
  float* pair   = (float*)alloc((size_t)LL * LL * HH * 4);
  float* logits = (float*)alloc((size_t)HH * LL * LL * 4);
  f16*   alpha16= (f16*)alloc((size_t)16 * LL * LL * 2);    // padded to 16 heads for A reads
  float* aggr   = (float*)alloc((size_t)LL * HP3 * 4);
  float* featall= (float*)alloc((size_t)LL * DOUT * 4);
  f16*   feat16 = (f16*)alloc((size_t)LL * DOUT * 2);
  float* attn   = (float*)alloc((size_t)LL * FF * 4);
  float* x1     = (float*)alloc((size_t)LL * FF * 4);
  f16*   x116   = (f16*)alloc((size_t)LL * FF * 2);
  f16*   h116   = (f16*)alloc((size_t)LL * FF * 2);
  f16*   h216   = (f16*)alloc((size_t)LL * FF * 2);
  float* h3     = (float*)alloc((size_t)LL * FF * 4);
  (void)ws_size; (void)n_in; (void)in_sizes; (void)out_size;

  const float SQ13 = 0.57735026919f;          // sqrt(1/3)
  const float SNODE = SQ13 / 5.656854249f;    // sqrt(1/3)/sqrt(32)

  // ---- f16 conversions / weight transposes ----
  k_conv16<<<(LL * FF + 255) / 256, 256, 0, stream>>>(x, (long long)LL * FF, x16, 1.f);
  k_tr16<<<(HQK * FF + 255) / 256, 256, 0, stream>>>(Wq, FF, HQK, WqT, HQK);
  k_tr16<<<(HQK * FF + 255) / 256, 256, 0, stream>>>(Wk, FF, HQK, WkT, HQK);
  k_tr16<<<(HQK * FF + 255) / 256, 256, 0, stream>>>(Wv, FF, HQK, WvT, HQK);
  k_tr16<<<(HP3 * FF + 255) / 256, 256, 0, stream>>>(Wqp, FF, HP3, WqpT, HP3);
  k_tr16<<<(HP3 * FF + 255) / 256, 256, 0, stream>>>(Wkp, FF, HP3, WkpT, HP3);
  k_tr16<<<(HP3 * FF + 255) / 256, 256, 0, stream>>>(Wvp, FF, HP3, WvpT, HP3);
  k_tr16<<<(16 * CC + 255) / 256, 256, 0, stream>>>(Wpb, CC, HH, WpbT, 16);
  k_tr16<<<(FF * DOUT + 255) / 256, 256, 0, stream>>>(Wout, DOUT, FF, WoutT, FF);
  k_tr16<<<(FF * FF + 255) / 256, 256, 0, stream>>>(W1, FF, FF, W1T, FF);
  k_tr16<<<(FF * FF + 255) / 256, 256, 0, stream>>>(W2, FF, FF, W2T, FF);
  k_tr16<<<(FF * FF + 255) / 256, 256, 0, stream>>>(W3, FF, FF, W3T, FF);
  k_zconv<<<((long long)LL * LL * CC + 255) / 256, 256, 0, stream>>>(z, z16, zt16);

  // ---- projections (WMMA) ----
  k_gemm<<<dim3(LL/16, HQK/16, 1), 32, 0, stream>>>(x16, FF, 0, WqT, FF, 0, nullptr, q16s, HQK, 0,
                                                    FF, LL, HQK, nullptr, SNODE, GF_OUTF16);
  k_gemm<<<dim3(LL/16, HQK/16, 1), 32, 0, stream>>>(x16, FF, 0, WkT, FF, 0, nullptr, k16, HQK, 0,
                                                    FF, LL, HQK, nullptr, 1.f, GF_OUTF16);
  k_gemm<<<dim3(LL/16, HQK/16, 1), 32, 0, stream>>>(x16, FF, 0, WvT, FF, 0, nullptr, vt16, LL, 0,
                                                    FF, LL, HQK, nullptr, 1.f, GF_OUTF16T);
  k_gemm<<<dim3(LL/16, HP3/16, 1), 32, 0, stream>>>(x16, FF, 0, WqpT, FF, 0, qp_l, nullptr, HP3, 0,
                                                    FF, LL, HP3, nullptr, 1.f, 0);
  k_gemm<<<dim3(LL/16, HP3/16, 1), 32, 0, stream>>>(x16, FF, 0, WkpT, FF, 0, kp_l, nullptr, HP3, 0,
                                                    FF, LL, HP3, nullptr, 1.f, 0);
  k_gemm<<<dim3(LL/16, HP3/16, 1), 32, 0, stream>>>(x16, FF, 0, WvpT, FF, 0, vp_l, nullptr, HP3, 0,
                                                    FF, LL, HP3, nullptr, 1.f, 0);
  // pair bias: (L*L x 64) @ (64 x 12), scale sqrt(1/3)
  k_gemm<<<dim3((LL*LL)/16, 1, 1), 32, 0, stream>>>(z16, CC, 0, WpbT, CC, 0, pair, nullptr, HH, 0,
                                                    CC, LL * LL, HH, nullptr, SQ13, 0);

  // ---- rigid transforms, norms, per-head coefficients ----
  k_points<<<(LL * HH + 255) / 256, 256, 0, stream>>>(R, t, qp_l, kp_l, vp_l, spc,
                                                      qpg, kpg, vpt16, qn, kn, gcoef);

  // ---- fused logits (2 WMMAs per tile) + mask ----
  k_combine<<<dim3(LL/16, LL/16, HH), 32, 0, stream>>>(q16s, k16, qpg, kpg, pair, qn, kn, gcoef,
                                                       mask, logits);
  // ---- softmax -> f16 alpha ----
  k_softmax<<<dim3(LL, HH), 128, 0, stream>>>(logits, mask, alpha16);

  // ---- aggregations (WMMA, batched over heads / residues) ----
  // feat_node[i][768 + h*32 + d] = sum_j alpha[h][i][j] * v[j][h*32+d]
  k_gemm<<<dim3(LL/16, VD/16, HH), 32, 0, stream>>>(alpha16, LL, (long long)LL * LL,
                                                    vt16, LL, (long long)32 * LL,
                                                    featall + 768, nullptr, DOUT, 32,
                                                    LL, LL, VD, nullptr, 1.f, 0);
  // aggr[i][h*24 + pc] = sum_j alpha[h][i][j] * vp_g[j][h][pc]
  k_gemm<<<dim3(LL/16, 2, HH), 32, 0, stream>>>(alpha16, LL, (long long)LL * LL,
                                                vpt16, LL, (long long)32 * LL,
                                                aggr, nullptr, HP3, 24,
                                                LL, LL, 24, nullptr, 1.f, 0);
  // feat_p2n[i][h*64 + c] = sum_j alpha[h][i][j] * z[i][j][c]  (batched over i)
  k_gemm<<<dim3(1, CC/16, LL), 32, 0, stream>>>(alpha16, LL * LL, (long long)LL,
                                                zt16, LL, (long long)CC * LL,
                                                featall, nullptr, CC, DOUT,
                                                LL, HH, CC, nullptr, 1.f, 0);
  // spatial features (local frame / dist / direction)
  k_spatial<<<(LL * HH * PVN + 255) / 256, 256, 0, stream>>>(R, t, aggr, featall);

  // ---- output projection + residual/LN + MLP + final LN ----
  k_conv16<<<((long long)LL * DOUT + 255) / 256, 256, 0, stream>>>(featall, (long long)LL * DOUT, feat16, 1.f);
  k_gemm<<<dim3(LL/16, FF/16, 1), 32, 0, stream>>>(feat16, DOUT, 0, WoutT, DOUT, 0,
                                                   attn, nullptr, FF, 0,
                                                   DOUT, LL, FF, bout, 1.f, 0);
  k_ln<<<LL, 128, 0, stream>>>(x, attn, mask, ln1g, ln1b, x1, x116);
  k_gemm<<<dim3(LL/16, FF/16, 1), 32, 0, stream>>>(x116, FF, 0, W1T, FF, 0, nullptr, h116, FF, 0,
                                                   FF, LL, FF, b1, 1.f, GF_OUTF16 | GF_RELU);
  k_gemm<<<dim3(LL/16, FF/16, 1), 32, 0, stream>>>(h116, FF, 0, W2T, FF, 0, nullptr, h216, FF, 0,
                                                   FF, LL, FF, b2, 1.f, GF_OUTF16 | GF_RELU);
  k_gemm<<<dim3(LL/16, FF/16, 1), 32, 0, stream>>>(h216, FF, 0, W3T, FF, 0, h3, nullptr, FF, 0,
                                                   FF, LL, FF, b3, 1.f, 0);
  k_ln<<<LL, 128, 0, stream>>>(x1, h3, nullptr, ln2g, ln2b, (float*)d_out, nullptr);
}
